// Decoder_7730941133123
// MI455X (gfx1250) — matrix operand order repaired and checked by reading, not compiled, not run on hardware
//
#include <hip/hip_runtime.h>
#include <hip/hip_bf16.h>
#include <math.h>

// Problem dims (from reference)
#define Bb 64
#define Ss 400
#define Hh 256
#define Ee 128
#define Vv 50000
#define Zz 500
#define Tt 50
#define H2 (2*Hh)          // 512
#define VZ (Vv+Zz)         // 50500

typedef __attribute__((ext_vector_type(16))) __bf16 v16bf;
typedef __attribute__((ext_vector_type(8)))  float  v8f;

// Branch-free activations on the hardware transcendental unit (v_exp_f32).
// Keeps EXEC all-ones around WMMA (libm tanhf/expf emit saveexec branches).
__device__ __forceinline__ float fast_tanh(float x) {
  const float e = __expf(2.f * x);
  return 1.f - 2.f / (e + 1.f);      // -> -1 / +1 at the extremes, no branches
}
__device__ __forceinline__ float fast_sigmoid(float x) {
  return 1.f / (1.f + __expf(-x));
}

// ---------------------------------------------------------------------------
// K0: context projection + LSTM cell (small, VALU). One block per batch row.
// ---------------------------------------------------------------------------
__global__ void ctx_lstm_kernel(const float* __restrict__ x_t,
                                const float* __restrict__ cte,
                                const float* __restrict__ s_h,
                                const float* __restrict__ s_c,
                                const float* __restrict__ W_ctx,
                                const float* __restrict__ b_ctx,
                                const float* __restrict__ w_ih,
                                const float* __restrict__ w_hh,
                                const float* __restrict__ b_ih,
                                const float* __restrict__ b_hh,
                                float* __restrict__ out_h,
                                float* __restrict__ out_c,
                                float* __restrict__ ws_t,
                                float* __restrict__ ws_sthat) {
  const int b = blockIdx.x, tid = threadIdx.x;   // 256 threads
  __shared__ float xin[Ee + H2];  // [x_t, cte] = 640
  __shared__ float tl[Ee];
  __shared__ float shl[Hh];
  for (int k = tid; k < Ee; k += 256) xin[k] = x_t[b*Ee + k];
  for (int k = tid; k < H2; k += 256) xin[Ee + k] = cte[b*H2 + k];
  shl[tid] = s_h[b*Hh + tid];
  __syncthreads();
  if (tid < Ee) {
    float acc = b_ctx[tid];
    const float* wr = W_ctx + (size_t)tid * (Ee + H2);
    for (int k = 0; k < Ee + H2; ++k) acc += xin[k] * wr[k];
    tl[tid] = acc;
    ws_t[b*Ee + tid] = acc;
  }
  __syncthreads();
  float gates[4];
  #pragma unroll
  for (int q = 0; q < 4; ++q) {
    const int row = q*Hh + tid;
    float acc = b_ih[row] + b_hh[row];
    const float* wi = w_ih + (size_t)row * Ee;
    for (int k = 0; k < Ee; ++k) acc += tl[k] * wi[k];
    const float* wh = w_hh + (size_t)row * Hh;
    for (int k = 0; k < Hh; ++k) acc += shl[k] * wh[k];
    gates[q] = acc;
  }
  const float ig = fast_sigmoid(gates[0]);
  const float fg = fast_sigmoid(gates[1]);
  const float gg = fast_tanh(gates[2]);
  const float og = fast_sigmoid(gates[3]);
  const float c = fg * s_c[b*Hh + tid] + ig * gg;
  const float h = og * fast_tanh(c);
  out_h[b*Hh + tid] = h;
  out_c[b*Hh + tid] = c;
  ws_sthat[b*H2 + tid]      = h;   // st_hat = [h, c]
  ws_sthat[b*H2 + Hh + tid] = c;
}

// ---------------------------------------------------------------------------
// Generic per-batch linear: out[b,c] = bias[c] + dot(A[b,:K], W[c,:K]). K<=512.
// ---------------------------------------------------------------------------
__global__ void linear_kernel(const float* __restrict__ A,
                              const float* __restrict__ W,
                              const float* __restrict__ bias,
                              float* __restrict__ out, int K, int C) {
  const int b = blockIdx.x;
  __shared__ float al[H2];
  for (int k = threadIdx.x; k < K; k += blockDim.x) al[k] = A[(size_t)b*K + k];
  __syncthreads();
  for (int c = threadIdx.x; c < C; c += blockDim.x) {
    float acc = bias[c];
    const float* wr = W + (size_t)c * K;
    for (int k = 0; k < K; ++k) acc += al[k] * wr[k];
    out[(size_t)b * C + c] = acc;
  }
}

// ---------------------------------------------------------------------------
// WMMA fused attention logits:
//   et[row] = sum_col v[col] * tanh( (A @ W^T)[row,col] + dech[b,col] )
// Block = 256 threads (8 waves) per 16-row tile. The A tile is staged ONCE
// into LDS as bf16 (16 x K <= 16KB), so global A traffic is a single pass
// (roofline minimum). Waves split the column tiles (stride 8*16), each runs
// the K-loop of v_wmma_f32_16x16x32_bf16 with A fragments from LDS and W
// fragments converted from global; tanh . v is applied in-register and the
// per-row sums are reduced via half-wave shuffles + an LDS cross-wave reduce.
// ---------------------------------------------------------------------------
__global__ void attn_logits_wmma(const float* __restrict__ A,
                                 const float* __restrict__ W,
                                 const float* __restrict__ dech,
                                 const float* __restrict__ vvec,
                                 float* __restrict__ et,
                                 int perB, int C, int K) {
  const int tid  = threadIdx.x;          // 256
  const int lane = tid & 31;
  const int wave = tid >> 5;             // 0..7
  const int g  = lane >> 4;              // half-wave group
  const int ln = lane & 15;
  const int tileRow = blockIdx.x * 16;

  __shared__ __bf16 ldsA[16 * H2];       // 16 rows x K (K <= 512), bf16
  __shared__ float  ldsRed[8][16];       // per-wave row partials

  // cooperative stage: A tile -> bf16 LDS (one global pass over A)
  for (int r = 0; r < 16; ++r)
    for (int k = tid; k < K; k += 256)
      ldsA[r*K + k] = (__bf16)A[(size_t)(tileRow + r) * K + k];
  __syncthreads();

  float rowAcc[8];
  #pragma unroll
  for (int r = 0; r < 8; ++r) rowAcc[r] = 0.f;

  for (int ct = wave * 16; ct < C; ct += 8 * 16) {
    v8f acc = {};
    for (int kk = 0; kk < K; kk += 32) {
      v16bf a, bm;
      const __bf16* arow = ldsA + ln * K + kk;                 // A row = ln
      const float*  wrow = W + (size_t)(ct + ln) * K + kk;     // B[k][n]=W[n][k]
      #pragma unroll
      for (int p = 0; p < 8; ++p) {
        const int k0 = ((p < 4) ? 2*p : 16 + 2*(p-4)) + 8*g;   // A 16x32 layout
        a[2*p]   = arow[k0];
        a[2*p+1] = arow[k0 + 1];
        const int kb = k0;                                     // B 32x16 layout (same k order as A)
        bm[2*p]   = (__bf16)wrow[kb];
        bm[2*p+1] = (__bf16)wrow[kb + 1];
      }
      acc = __builtin_amdgcn_wmma_f32_16x16x32_bf16(
              false, a, false, bm, (short)0, acc, false, false);
    }
    // D(m,n): VGPR r <-> m = 8*g + r, lane <-> n; branch-free epilogue
    #pragma unroll
    for (int r = 0; r < 8; ++r) {
      const int grow = tileRow + 8*g + r;
      const int bidx = grow / perB;
      const int col  = ct + ln;
      rowAcc[r] += fast_tanh(acc[r] + dech[(size_t)bidx * C + col]) * vvec[col];
    }
  }
  // reduce over the 16 column-lanes of each half-wave, park per-wave partials
  #pragma unroll
  for (int r = 0; r < 8; ++r) {
    float v = rowAcc[r];
    v += __shfl_xor(v, 8, 32);
    v += __shfl_xor(v, 4, 32);
    v += __shfl_xor(v, 2, 32);
    v += __shfl_xor(v, 1, 32);
    if (ln == 0) ldsRed[wave][8*g + r] = v;
  }
  __syncthreads();
  if (tid < 16) {
    float v = 0.f;
    #pragma unroll
    for (int w = 0; w < 8; ++w) v += ldsRed[w][tid];
    et[tileRow + tid] = v;
  }
}

// ---------------------------------------------------------------------------
// K2: encoder temporal softmax + at normalization + cte_new einsum.
// ---------------------------------------------------------------------------
__global__ void enc_attn_finalize(const float* __restrict__ et,
                                  const float* __restrict__ sumT,
                                  const float* __restrict__ pad,
                                  const float* __restrict__ enc,
                                  float* __restrict__ out_sumT,
                                  float* __restrict__ out_cte,
                                  float* __restrict__ ws_at) {
  const int b = blockIdx.x, tid = threadIdx.x;  // 256 threads
  __shared__ float at[Ss];
  __shared__ float red[256];
  float part = 0.f;
  for (int s = tid; s < Ss; s += 256) {
    const float ex = __expf(et[b*Ss + s]);
    const float st = sumT[b*Ss + s];
    out_sumT[b*Ss + s] = st + ex;
    const float a = (ex / st) * pad[b*Ss + s];
    at[s] = a;
    part += a;
  }
  red[tid] = part; __syncthreads();
  for (int off = 128; off >= 1; off >>= 1) {
    if (tid < off) red[tid] += red[tid + off];
    __syncthreads();
  }
  const float inv = 1.f / red[0];
  for (int s = tid; s < Ss; s += 256) { at[s] *= inv; ws_at[b*Ss + s] = at[s]; }
  __syncthreads();
  for (int hc = tid; hc < H2; hc += 256) {       // coalesced over 2H
    float acc = 0.f;
    for (int s = 0; s < Ss; ++s)
      acc += at[s] * enc[((size_t)(b*Ss + s)) * H2 + hc];
    out_cte[b*H2 + hc] = acc;
  }
}

// ---------------------------------------------------------------------------
// K4: decoder softmax + ctd + generation gate + v1 hidden.
// ---------------------------------------------------------------------------
__global__ void dec_attn_gen_hidden(const float* __restrict__ etd,
                                    const float* __restrict__ prev_state,
                                    const float* __restrict__ out_h,
                                    const float* __restrict__ out_c,
                                    const float* __restrict__ out_cte,
                                    const float* __restrict__ ws_t,
                                    const float* __restrict__ fc_w,
                                    const float* __restrict__ fc_b,
                                    const float* __restrict__ v1_w,
                                    const float* __restrict__ v1_b,
                                    float* __restrict__ ws_ctd,
                                    float* __restrict__ ws_gen,
                                    float* __restrict__ ws_hidden) {
  const int b = blockIdx.x, tid = threadIdx.x;  // 256 threads
  __shared__ float atd[Tt];
  __shared__ float red[256];
  __shared__ float ctd_l[Hh];
  if (tid == 0) {
    float mx = -1e30f;
    for (int t = 0; t < Tt; ++t) mx = fmaxf(mx, etd[b*Tt + t]);
    float sum = 0.f;
    for (int t = 0; t < Tt; ++t) { const float e = __expf(etd[b*Tt + t] - mx); atd[t] = e; sum += e; }
    const float inv = 1.f / sum;
    for (int t = 0; t < Tt; ++t) atd[t] *= inv;
  }
  __syncthreads();
  {
    float acc = 0.f;
    for (int t = 0; t < Tt; ++t)
      acc += atd[t] * prev_state[((size_t)(b*Tt + t)) * Hh + tid];
    ctd_l[tid] = acc;
    ws_ctd[b*Hh + tid] = acc;
  }
  __syncthreads();
  // gen = sigmoid(fc_w . [cte_new(512), ctd(256), h(256), c(256), t(128)] + fc_b)
  float part = 0.f;
  for (int k = tid; k < 5*Hh + Ee; k += 256) {
    float xv;
    if      (k < 512)  xv = out_cte[b*H2 + k];
    else if (k < 768)  xv = ctd_l[k - 512];
    else if (k < 1024) xv = out_h[b*Hh + (k - 768)];
    else if (k < 1280) xv = out_c[b*Hh + (k - 1024)];
    else               xv = ws_t[b*Ee + (k - 1280)];
    part += xv * fc_w[k];
  }
  red[tid] = part; __syncthreads();
  for (int off = 128; off >= 1; off >>= 1) {
    if (tid < off) red[tid] += red[tid + off];
    __syncthreads();
  }
  const float gen = fast_sigmoid(red[0] + fc_b[0]);
  if (tid == 0) ws_gen[b] = gen;
  // hidden[j] = v1_b[j] + v1_w[j,:] . [h(256), cte_new(512), ctd(256)]
  float acc = v1_b[tid];
  const float* wr = v1_w + (size_t)tid * (4*Hh);
  for (int k = 0; k < Hh; ++k) acc += out_h[b*Hh + k] * wr[k];
  for (int k = 0; k < H2; ++k) acc += out_cte[b*H2 + k] * wr[Hh + k];
  for (int k = 0; k < Hh; ++k) acc += ctd_l[k] * wr[Hh + H2 + k];
  ws_hidden[b*Hh + tid] = acc;
}

// ---------------------------------------------------------------------------
// K5: WMMA vocab GEMM: logits[64,50000] = hidden[64,256] @ v2_w.T + v2_b.
// One wave per 16x16 tile; bound by streaming v2_w (51MB) once from HBM.
// ---------------------------------------------------------------------------
__global__ void vocab_logits_wmma(const float* __restrict__ Hmat,
                                  const float* __restrict__ V2,
                                  const float* __restrict__ v2b,
                                  float* __restrict__ logits) {
  const int lane = threadIdx.x;     // 32
  const int g = lane >> 4, ln = lane & 15;
  const int rbase = blockIdx.y * 16;
  const int cbase = blockIdx.x * 16;
  v8f acc = {};
  for (int kk = 0; kk < Hh; kk += 32) {
    v16bf a, bm;
    const float* arow = Hmat + (size_t)(rbase + ln) * Hh + kk;
    const float* wrow = V2   + (size_t)(cbase + ln) * Hh + kk;
    #pragma unroll
    for (int p = 0; p < 8; ++p) {
      const int k0 = ((p < 4) ? 2*p : 16 + 2*(p-4)) + 8*g;
      a[2*p]   = (__bf16)arow[k0];
      a[2*p+1] = (__bf16)arow[k0 + 1];
      const int kb = k0;
      bm[2*p]   = (__bf16)wrow[kb];
      bm[2*p+1] = (__bf16)wrow[kb + 1];
    }
    acc = __builtin_amdgcn_wmma_f32_16x16x32_bf16(
            false, a, false, bm, (short)0, acc, false, false);
  }
  #pragma unroll
  for (int r = 0; r < 8; ++r) {
    const int m = 8*g + r;
    logits[(size_t)(rbase + m) * Vv + cbase + ln] = acc[r] + v2b[cbase + ln];
  }
}

// ---------------------------------------------------------------------------
// K6: per-row max and sum-exp over V=50000.
// ---------------------------------------------------------------------------
__global__ void row_softmax_stats(const float* __restrict__ logits,
                                  float* __restrict__ rowmax,
                                  float* __restrict__ rowsum) {
  const int b = blockIdx.x, tid = threadIdx.x;
  __shared__ float red[256];
  float mx = -1e30f;
  for (int v = tid; v < Vv; v += 256) mx = fmaxf(mx, logits[(size_t)b*Vv + v]);
  red[tid] = mx; __syncthreads();
  for (int off = 128; off >= 1; off >>= 1) {
    if (tid < off) red[tid] = fmaxf(red[tid], red[tid + off]);
    __syncthreads();
  }
  mx = red[0]; __syncthreads();
  float sum = 0.f;
  for (int v = tid; v < Vv; v += 256) sum += __expf(logits[(size_t)b*Vv + v] - mx);
  red[tid] = sum; __syncthreads();
  for (int off = 128; off >= 1; off >>= 1) {
    if (tid < off) red[tid] += red[tid + off];
    __syncthreads();
  }
  if (tid == 0) { rowmax[b] = mx; rowsum[b] = red[0]; }
}

// ---------------------------------------------------------------------------
// K7: dist = gen * softmax(logits), OOV tail zeroed.
// ---------------------------------------------------------------------------
__global__ void write_dist(const float* __restrict__ logits,
                           const float* __restrict__ rowmax,
                           const float* __restrict__ rowsum,
                           const float* __restrict__ gen,
                           float* __restrict__ dist) {
  const size_t i = (size_t)blockIdx.x * blockDim.x + threadIdx.x;
  if (i >= (size_t)Bb * VZ) return;
  const int b = (int)(i / VZ);
  const int j = (int)(i % VZ);
  float d = 0.f;
  if (j < Vv)
    d = gen[b] * __expf(logits[(size_t)b*Vv + j] - rowmax[b]) / rowsum[b];
  dist[i] = d;
}

// ---------------------------------------------------------------------------
// K8: pointer scatter-add: dist[b, idx[b,s]] += (1-gen[b]) * at[b,s].
// ---------------------------------------------------------------------------
__global__ void scatter_pointer(const int* __restrict__ idx,
                                const float* __restrict__ ws_at,
                                const float* __restrict__ gen,
                                float* __restrict__ dist) {
  const int i = blockIdx.x * blockDim.x + threadIdx.x;
  if (i >= Bb * Ss) return;
  const int b = i / Ss;
  atomicAdd(&dist[(size_t)b * VZ + idx[i]], (1.f - gen[b]) * ws_at[i]);
}

// ---------------------------------------------------------------------------
// K9: prev_state_new = concat(prev_state, h[:,None,:]).
// ---------------------------------------------------------------------------
__global__ void prev_state_append(const float* __restrict__ prev,
                                  const float* __restrict__ h,
                                  float* __restrict__ out) {
  const size_t i = (size_t)blockIdx.x * blockDim.x + threadIdx.x;
  if (i >= (size_t)Bb * (Tt+1) * Hh) return;
  const int b   = (int)(i / ((Tt+1)*Hh));
  const int rem = (int)(i % ((Tt+1)*Hh));
  const int t   = rem / Hh;
  const int hc  = rem % Hh;
  out[i] = (t < Tt) ? prev[((size_t)b*Tt + t)*Hh + hc] : h[b*Hh + hc];
}

// ---------------------------------------------------------------------------
extern "C" void kernel_launch(void* const* d_in, const int* in_sizes, int n_in,
                              void* d_out, int out_size, void* d_ws, size_t ws_size,
                              hipStream_t stream) {
  (void)in_sizes; (void)n_in; (void)out_size; (void)ws_size;
  const float* x_t   = (const float*)d_in[0];
  const float* s_h   = (const float*)d_in[1];
  const float* s_c   = (const float*)d_in[2];
  const float* enc   = (const float*)d_in[3];
  const float* pad   = (const float*)d_in[4];
  const float* cte   = (const float*)d_in[5];
  /* zeros d_in[6] unused: tail written directly */
  const float* sumT  = (const float*)d_in[7];
  const float* prev  = (const float*)d_in[8];
  const int*   ebx   = (const int*)  d_in[9];
  const float* W_ctx = (const float*)d_in[10];
  const float* b_ctx = (const float*)d_in[11];
  const float* w_ih  = (const float*)d_in[12];
  const float* w_hh  = (const float*)d_in[13];
  const float* b_ih  = (const float*)d_in[14];
  const float* b_hh  = (const float*)d_in[15];
  const float* enc_wh   = (const float*)d_in[16];
  const float* enc_ws   = (const float*)d_in[17];
  const float* enc_ws_b = (const float*)d_in[18];
  const float* enc_v    = (const float*)d_in[19];
  const float* dec_wprev= (const float*)d_in[20];
  const float* dec_ws   = (const float*)d_in[21];
  const float* dec_ws_b = (const float*)d_in[22];
  const float* dec_v    = (const float*)d_in[23];
  const float* fc_w  = (const float*)d_in[24];
  const float* fc_b  = (const float*)d_in[25];
  const float* v1_w  = (const float*)d_in[26];
  const float* v1_b  = (const float*)d_in[27];
  const float* v2_w  = (const float*)d_in[28];
  const float* v2_b  = (const float*)d_in[29];

  // d_out layout (tuple flattened): dist, h, c, cte_new, sum_temporal_new, prev_state_new
  float* out = (float*)d_out;
  float* o_dist = out;                                  // 64*50500
  float* o_h    = o_dist + (size_t)Bb * VZ;             // 64*256
  float* o_c    = o_h    + (size_t)Bb * Hh;             // 64*256
  float* o_cte  = o_c    + (size_t)Bb * Hh;             // 64*512
  float* o_sumT = o_cte  + (size_t)Bb * H2;             // 64*400
  float* o_prev = o_sumT + (size_t)Bb * Ss;             // 64*51*256

  // workspace layout (floats)
  float* w = (float*)d_ws;
  float* ws_t      = w;                      w += (size_t)Bb * Ee;
  float* ws_sthat  = w;                      w += (size_t)Bb * H2;
  float* ws_dech   = w;                      w += (size_t)Bb * H2;
  float* ws_decd   = w;                      w += (size_t)Bb * Hh;
  float* ws_et     = w;                      w += (size_t)Bb * Ss;
  float* ws_at     = w;                      w += (size_t)Bb * Ss;
  float* ws_etd    = w;                      w += (size_t)Bb * Tt;
  float* ws_ctd    = w;                      w += (size_t)Bb * Hh;
  float* ws_gen    = w;                      w += (size_t)Bb;
  float* ws_hidden = w;                      w += (size_t)Bb * Hh;
  float* ws_rowmax = w;                      w += (size_t)Bb;
  float* ws_rowsum = w;                      w += (size_t)Bb;
  float* ws_logits = w;                      /* 64*50000 */

  // 1) context projection + LSTM
  ctx_lstm_kernel<<<Bb, 256, 0, stream>>>(x_t, cte, s_h, s_c, W_ctx, b_ctx,
                                          w_ih, w_hh, b_ih, b_hh,
                                          o_h, o_c, ws_t, ws_sthat);
  // 2) dech = st_hat @ enc_ws.T + b ; decd = h @ dec_ws.T + b
  linear_kernel<<<Bb, 256, 0, stream>>>(ws_sthat, enc_ws, enc_ws_b, ws_dech, H2, H2);
  linear_kernel<<<Bb, 256, 0, stream>>>(o_h, dec_ws, dec_ws_b, ws_decd, Hh, Hh);
  // 3) encoder attention logits (WMMA, fused tanh . v): [25600,512]x[512,512]
  attn_logits_wmma<<<(Bb*Ss)/16, 256, 0, stream>>>(enc, enc_wh, ws_dech, enc_v,
                                                   ws_et, Ss, H2, H2);
  // 4) encoder temporal softmax + cte_new
  enc_attn_finalize<<<Bb, 256, 0, stream>>>(ws_et, sumT, pad, enc,
                                            o_sumT, o_cte, ws_at);
  // 5) decoder attention logits (WMMA): [3200,256]x[256,256]
  attn_logits_wmma<<<(Bb*Tt)/16, 256, 0, stream>>>(prev, dec_wprev, ws_decd, dec_v,
                                                   ws_etd, Tt, Hh, Hh);
  // 6) decoder softmax + ctd + gen + v1 hidden
  dec_attn_gen_hidden<<<Bb, 256, 0, stream>>>(ws_etd, prev, o_h, o_c, o_cte, ws_t,
                                              fc_w, fc_b, v1_w, v1_b,
                                              ws_ctd, ws_gen, ws_hidden);
  // 7) vocab logits (WMMA): [64,256]x[256,50000]
  vocab_logits_wmma<<<dim3(Vv/16, Bb/16), 32, 0, stream>>>(ws_hidden, v2_w, v2_b,
                                                           ws_logits);
  // 8) softmax stats, dist write, pointer scatter
  row_softmax_stats<<<Bb, 256, 0, stream>>>(ws_logits, ws_rowmax, ws_rowsum);
  write_dist<<<(int)(((size_t)Bb*VZ + 255)/256), 256, 0, stream>>>(ws_logits,
                                       ws_rowmax, ws_rowsum, ws_gen, o_dist);
  scatter_pointer<<<(Bb*Ss + 255)/256, 256, 0, stream>>>(ebx, ws_at, ws_gen, o_dist);
  // 9) prev_state_new
  prev_state_append<<<(int)(((size_t)Bb*(Tt+1)*Hh + 255)/256), 256, 0, stream>>>(
      prev, o_h, o_prev);
}
